// MultiHeadAttention_38912403702183
// MI455X (gfx1250) — matrix operand order repaired and checked by reading, not compiled, not run on hardware
//
#include <hip/hip_runtime.h>
#include <hip/hip_bf16.h>

typedef __attribute__((ext_vector_type(16))) _Float16 v16h;
typedef __attribute__((ext_vector_type(8)))  _Float16 v8h;
typedef __attribute__((ext_vector_type(8)))  float    v8f;

#define DMODEL 1024
#define NHEADS 16
#define DK     64
#define BATCH  4
#define SEQ    2048
#define MROWS  (BATCH * SEQ)

__device__ __forceinline__ v8f wmma_f16(v16h a, v16h b, v8f c) {
  // D = A(16x32 f16) * B(32x16 f16) + C(16x16 f32)
  return __builtin_amdgcn_wmma_f32_16x16x32_f16(false, a, false, b, (short)0, c,
                                                false, false);
}

// Async copy 16 bytes global -> LDS (ASYNCcnt-tracked, no VGPR round trip).
__device__ __forceinline__ void async_b128(void* lds, const void* g) {
  unsigned l = (unsigned)(size_t)lds;                 // LDS byte offset
  unsigned long long ga = (unsigned long long)(size_t)g;
  asm volatile("global_load_async_to_lds_b128 %0, %1, off"
               :: "v"(l), "v"(ga) : "memory");
}

__device__ __forceinline__ void wait_async0() {
#if __has_builtin(__builtin_amdgcn_s_wait_asynccnt)
  __builtin_amdgcn_s_wait_asynccnt(0);
#else
  asm volatile("s_wait_asynccnt 0" ::: "memory");
#endif
}

// ---------------------------------------------------------------------------
// fp32 -> fp16 conversion (vectorized x4)
// ---------------------------------------------------------------------------
__global__ void cvt_f32_f16(const float* __restrict__ in,
                            _Float16* __restrict__ out, int n) {
  int i = (blockIdx.x * blockDim.x + threadIdx.x) * 4;
  if (i + 3 < n) {
    float4 v = *(const float4*)(in + i);
    out[i + 0] = (_Float16)v.x;
    out[i + 1] = (_Float16)v.y;
    out[i + 2] = (_Float16)v.z;
    out[i + 3] = (_Float16)v.w;
  } else {
    for (; i < n; ++i) out[i] = (_Float16)in[i];
  }
}

// ---------------------------------------------------------------------------
// Tiled WMMA GEMM: C[M,N] = A[M,K] * W[K,N] + bias[N]
// 128 threads (4 waves), 64x64 output tile, each wave a 32x32 subtile
// (2x2 WMMA accumulators), K stepped by 32. Tiles staged in LDS via
// global_load_async_to_lds_b128.
// ---------------------------------------------------------------------------
template <bool SPLIT, bool OUTF32>
__global__ __launch_bounds__(128) void gemm_wmma(
    const _Float16* __restrict__ A, const _Float16* __restrict__ W,
    const float* __restrict__ bias, _Float16* __restrict__ outh,
    float* __restrict__ outf, int M, int N, int K) {
  __shared__ __align__(16) _Float16 As[64][40];  // 64x32 tile (+pad, 80B rows)
  __shared__ __align__(16) _Float16 Bs[32][72];  // 32x64 tile (+pad, 144B rows)

  const int t = threadIdx.x;
  const int lane = t & 31, wave = t >> 5;
  const int grp = lane >> 4, l16 = lane & 15;
  const int m0 = blockIdx.y * 64, n0 = blockIdx.x * 64;
  const int wm = (wave >> 1) * 32, wn = (wave & 1) * 32;

  const int arow = t >> 1, aseg = t & 1;   // A tile: 16 halves/thread
  const int brow = t >> 2, bseg = t & 3;   // B tile: 16 halves/thread

  v8f acc[2][2] = {};

  for (int kb = 0; kb < K; kb += 32) {
    {  // async A tile 64x32
      const _Float16* src = A + (size_t)(m0 + arow) * K + kb + aseg * 16;
      async_b128(&As[arow][aseg * 16], src);
      async_b128(&As[arow][aseg * 16 + 8], src + 8);
    }
    {  // async B tile 32x64
      const _Float16* src = W + (size_t)(kb + brow) * N + n0 + bseg * 16;
      async_b128(&Bs[brow][bseg * 16], src);
      async_b128(&Bs[brow][bseg * 16 + 8], src + 8);
    }
    if (kb + 32 < K) {  // prefetch next K-step tiles into L2
      __builtin_prefetch(A + (size_t)(m0 + arow) * K + kb + 32 + aseg * 16, 0, 3);
      __builtin_prefetch(W + (size_t)(kb + 32 + brow) * N + n0 + bseg * 16, 0, 3);
    }
    wait_async0();
    __syncthreads();

    // A fragments: lane row = M, elements: k = grp*8 + (e%8) + (e/8)*16
    v16h afr[2], bfr[2];
#pragma unroll
    for (int mi = 0; mi < 2; ++mi) {
      int row = wm + mi * 16 + l16;
      union { v16h v; v8h h[2]; } u;
      u.h[0] = *(const v8h*)&As[row][grp * 8];
      u.h[1] = *(const v8h*)&As[row][grp * 8 + 16];
      afr[mi] = u.v;
    }
    // B fragments: lane col = N, elements: k = grp*8 + (e%8) + (e/8)*16 (column read)
#pragma unroll
    for (int ni = 0; ni < 2; ++ni) {
      int col = wn + ni * 16 + l16;
      union { v16h v; _Float16 e[16]; } u;
#pragma unroll
      for (int e = 0; e < 16; ++e) u.e[e] = Bs[grp * 8 + (e & 7) + (e >> 3) * 16][col];
      bfr[ni] = u.v;
    }
#pragma unroll
    for (int mi = 0; mi < 2; ++mi)
#pragma unroll
      for (int ni = 0; ni < 2; ++ni)
        acc[mi][ni] = wmma_f16(afr[mi], bfr[ni], acc[mi][ni]);
    __syncthreads();
  }

  // Epilogue: C/D layout -> element r is row (grp*8+r), column l16
#pragma unroll
  for (int mi = 0; mi < 2; ++mi) {
#pragma unroll
    for (int ni = 0; ni < 2; ++ni) {
      int col = n0 + wn + ni * 16 + l16;
      float bv = bias[col];
#pragma unroll
      for (int r = 0; r < 8; ++r) {
        int row = m0 + wm + mi * 16 + grp * 8 + r;
        float v = acc[mi][ni][r] + bv;
        if (OUTF32) {
          outf[(size_t)row * N + col] = v;
        } else if (SPLIT) {
          int b = row / SEQ, s = row % SEQ, h = col >> 6, d = col & 63;
          outh[(((size_t)(b * NHEADS + h)) * SEQ + s) * DK + d] = (_Float16)v;
        } else {
          outh[(size_t)row * N + col] = (_Float16)v;
        }
      }
    }
  }
}

// ---------------------------------------------------------------------------
// Flash attention: grid = (SEQ/64, B*H), 128 threads (4 waves).
// Each wave owns 16 query rows; streams keys in blocks of 32 with online
// softmax. V key-block and mask byte-tile staged via async-to-LDS, overlapped
// with the QK^T WMMAs that read K straight from global (contiguous B-frags).
// ---------------------------------------------------------------------------
__global__ __launch_bounds__(128) void attn_flash(
    const _Float16* __restrict__ Q, const _Float16* __restrict__ Km,
    const _Float16* __restrict__ V, const unsigned char* __restrict__ mask,
    _Float16* __restrict__ ctx) {
  __shared__ __align__(16) _Float16 Vs[32][72];       // V key-block tile
  __shared__ __align__(16) _Float16 Ps[4][16][40];    // per-wave P (16x32)
  __shared__ __align__(16) unsigned char Ms[64][48];  // mask tile 64q x 32k

  const int t = threadIdx.x, lane = t & 31, wave = t >> 5;
  const int grp = lane >> 4, l16 = lane & 15;
  const int bh = blockIdx.y, b = bh >> 4, h = bh & 15;
  const int bq0 = blockIdx.x * 64;
  const int q0 = bq0 + wave * 16;
  const size_t base = (size_t)bh * SEQ * DK;

  const int vrow = t >> 2, vseg = t & 3;  // V tile: 16 halves/thread
  const int mrow_t = t >> 1, mhalf = t & 1;  // mask tile: 16 bytes/thread

  // Q A-fragments, loaded once: rows q0..q0+15, d split into two 32-chunks
  v16h aq[2];
  {
    const _Float16* qp = Q + base + (size_t)(q0 + l16) * DK;
#pragma unroll
    for (int dh = 0; dh < 2; ++dh) {
      union { v16h v; v8h h[2]; } u;
      u.h[0] = *(const v8h*)(qp + dh * 32 + grp * 16);
      u.h[1] = *(const v8h*)(qp + dh * 32 + grp * 16 + 8);
      aq[dh] = u.v;
    }
  }

  float mrow[8], lrow[8];
#pragma unroll
  for (int r = 0; r < 8; ++r) { mrow[r] = -3.0e38f; lrow[r] = 0.f; }
  v8f acc[4] = {};

  const unsigned char* mb = mask + (size_t)b * SEQ * SEQ;

  for (int kb = 0; kb < SEQ; kb += 32) {
    {  // async V key-block (32x64 halves) + mask tile (64x32 bytes)
      const _Float16* src = V + base + (size_t)(kb + vrow) * DK + vseg * 16;
      async_b128(&Vs[vrow][vseg * 16], src);
      async_b128(&Vs[vrow][vseg * 16 + 8], src + 8);
      async_b128(&Ms[mrow_t][mhalf * 16],
                 mb + (size_t)(bq0 + mrow_t) * SEQ + kb + mhalf * 16);
    }

    // scores = Q * K^T (overlaps with the async fills above):
    // K^T B-frag is a contiguous 16-half slice of a K row
    v8f sc[2];
#pragma unroll
    for (int kt = 0; kt < 2; ++kt) {
      v8f s = {};
#pragma unroll
      for (int dh = 0; dh < 2; ++dh) {
        int key = kb + kt * 16 + l16;
        v16h bk = *(const v16h*)(Km + base + (size_t)key * DK + dh * 32 + grp * 16);
        s = wmma_f16(aq[dh], bk, s);
      }
      sc[kt] = s;
    }

    wait_async0();
    __syncthreads();  // Vs + Ms ready

    // scale + mask + online softmax (row stats across 16-lane column groups)
#pragma unroll
    for (int r = 0; r < 8; ++r) {
      int ql = wave * 16 + grp * 8 + r;  // q-row within the 64-row block
      float s0 = sc[0][r] * 0.125f;      // 1/sqrt(64)
      float s1 = sc[1][r] * 0.125f;
      if (Ms[ql][l16]) s0 = -1e9f;
      if (Ms[ql][16 + l16]) s1 = -1e9f;
      float tmax = fmaxf(s0, s1);
      tmax = fmaxf(tmax, __shfl_xor(tmax, 1));
      tmax = fmaxf(tmax, __shfl_xor(tmax, 2));
      tmax = fmaxf(tmax, __shfl_xor(tmax, 4));
      tmax = fmaxf(tmax, __shfl_xor(tmax, 8));
      float mnew = fmaxf(mrow[r], tmax);
      float alpha = __expf(mrow[r] - mnew);
      float p0 = __expf(s0 - mnew), p1 = __expf(s1 - mnew);
      float rs = p0 + p1;
      rs += __shfl_xor(rs, 1);
      rs += __shfl_xor(rs, 2);
      rs += __shfl_xor(rs, 4);
      rs += __shfl_xor(rs, 8);
      lrow[r] = lrow[r] * alpha + rs;
      mrow[r] = mnew;
#pragma unroll
      for (int dt = 0; dt < 4; ++dt) acc[dt][r] *= alpha;
      Ps[wave][grp * 8 + r][l16] = (_Float16)p0;
      Ps[wave][grp * 8 + r][16 + l16] = (_Float16)p1;
    }
    __syncthreads();  // Ps visible (D-layout -> A-layout reshape)

    // ctx += P * V : P as A-frag from LDS, V block as B-frags from LDS
    v16h pfr;
    {
      union { v16h v; v8h h[2]; } u;
      u.h[0] = *(const v8h*)&Ps[wave][l16][grp * 8];
      u.h[1] = *(const v8h*)&Ps[wave][l16][grp * 8 + 16];
      pfr = u.v;
    }
#pragma unroll
    for (int dt = 0; dt < 4; ++dt) {
      union { v16h v; _Float16 e[16]; } u;
#pragma unroll
      for (int e = 0; e < 16; ++e) u.e[e] = Vs[grp * 8 + (e & 7) + (e >> 3) * 16][dt * 16 + l16];
      acc[dt] = wmma_f16(pfr, u.v, acc[dt]);
    }
    __syncthreads();  // before Vs/Ms are overwritten next iteration
  }

  // normalize and write ctx in [B,S,H*DK] (merged-head) layout
#pragma unroll
  for (int r = 0; r < 8; ++r) {
    int q = q0 + grp * 8 + r;
    float inv = 1.f / lrow[r];
    _Float16* cp = ctx + ((size_t)b * SEQ + q) * DMODEL + h * DK;
#pragma unroll
    for (int dt = 0; dt < 4; ++dt)
      cp[dt * 16 + l16] = (_Float16)(acc[dt][r] * inv);
  }
}

// ---------------------------------------------------------------------------
extern "C" void kernel_launch(void* const* d_in, const int* in_sizes, int n_in,
                              void* d_out, int out_size, void* d_ws,
                              size_t ws_size, hipStream_t stream) {
  const float* x = (const float*)d_in[0];
  const unsigned char* mask = (const unsigned char*)d_in[1];
  const float* Wq = (const float*)d_in[2];
  const float* bq = (const float*)d_in[3];
  const float* Wk = (const float*)d_in[4];
  const float* bk = (const float*)d_in[5];
  const float* Wv = (const float*)d_in[6];
  const float* bv = (const float*)d_in[7];
  const float* Wo = (const float*)d_in[8];
  const float* bo = (const float*)d_in[9];
  float* out = (float*)d_out;

  // f16 workspace layout
  _Float16* ws = (_Float16*)d_ws;
  size_t o = 0;
  _Float16* xh  = ws + o; o += (size_t)MROWS * DMODEL;
  _Float16* wqh = ws + o; o += (size_t)DMODEL * DMODEL;
  _Float16* wkh = ws + o; o += (size_t)DMODEL * DMODEL;
  _Float16* wvh = ws + o; o += (size_t)DMODEL * DMODEL;
  _Float16* woh = ws + o; o += (size_t)DMODEL * DMODEL;
  _Float16* Qh  = ws + o; o += (size_t)MROWS * DMODEL;
  _Float16* Kh  = ws + o; o += (size_t)MROWS * DMODEL;
  _Float16* Vh  = ws + o; o += (size_t)MROWS * DMODEL;
  _Float16* ch  = ws + o; o += (size_t)MROWS * DMODEL;

  const int nx = MROWS * DMODEL;
  const int nw = DMODEL * DMODEL;
  cvt_f32_f16<<<(nx / 4 + 255) / 256, 256, 0, stream>>>(x, xh, nx);
  cvt_f32_f16<<<(nw / 4 + 255) / 256, 256, 0, stream>>>(Wq, wqh, nw);
  cvt_f32_f16<<<(nw / 4 + 255) / 256, 256, 0, stream>>>(Wk, wkh, nw);
  cvt_f32_f16<<<(nw / 4 + 255) / 256, 256, 0, stream>>>(Wv, wvh, nw);
  cvt_f32_f16<<<(nw / 4 + 255) / 256, 256, 0, stream>>>(Wo, woh, nw);

  dim3 gg(DMODEL / 64, MROWS / 64);
  gemm_wmma<true, false><<<gg, 128, 0, stream>>>(xh, wqh, bq, Qh, nullptr,
                                                 MROWS, DMODEL, DMODEL);
  gemm_wmma<true, false><<<gg, 128, 0, stream>>>(xh, wkh, bk, Kh, nullptr,
                                                 MROWS, DMODEL, DMODEL);
  gemm_wmma<true, false><<<gg, 128, 0, stream>>>(xh, wvh, bv, Vh, nullptr,
                                                 MROWS, DMODEL, DMODEL);

  dim3 ga(SEQ / 64, BATCH * NHEADS);
  attn_flash<<<ga, 128, 0, stream>>>(Qh, Kh, Vh, mask, ch);

  gemm_wmma<false, true><<<gg, 128, 0, stream>>>(ch, woh, bo, nullptr, out,
                                                 MROWS, DMODEL, DMODEL);
}
